// QwenDoubleStreamAttention_6253472383567
// MI455X (gfx1250) — matrix operand order repaired and checked by reading, not compiled, not run on hardware
//
#include <hip/hip_runtime.h>

#define DIM   3072
#define HEADS 24
#define HD    128
#define S_TXT 512
#define S_IMG 2048
#define S_TOT 2560
#define MT    128      // GEMM block tile M
#define NT    64       // GEMM block tile N
#define KS    32       // GEMM k-step

typedef __attribute__((ext_vector_type(16))) __bf16 bf16x16;
typedef __attribute__((ext_vector_type(8)))  float  f32x8;
typedef __attribute__((ext_vector_type(4)))  int    v4i;

union Frag16 {
  bf16x16 v;
  unsigned int ui[8];
  unsigned short us[16];
};

__device__ __forceinline__ unsigned short f2bf(float f) {
  unsigned int u = __float_as_uint(f);
  u += 0x7FFFu + ((u >> 16) & 1u);   // round-to-nearest-even
  return (unsigned short)(u >> 16);
}
__device__ __forceinline__ unsigned int pack2(float a, float b) {
  return (unsigned int)f2bf(a) | ((unsigned int)f2bf(b) << 16);
}

// ---- CDNA5 async global->LDS copy (ASYNCcnt) with safe fallback ----------
#ifndef USE_ASYNC_LDS
#if defined(__has_builtin)
#if __has_builtin(__builtin_amdgcn_global_load_async_to_lds_b128) && \
    __has_builtin(__builtin_amdgcn_s_wait_asynccnt)
#define USE_ASYNC_LDS 1
#endif
#endif
#endif
#ifndef USE_ASYNC_LDS
#define USE_ASYNC_LDS 0
#endif

__device__ __forceinline__ void cp16(const unsigned short* g, unsigned short* l) {
#if USE_ASYNC_LDS
  __builtin_amdgcn_global_load_async_to_lds_b128(
      (__attribute__((address_space(1))) v4i*)g,
      (__attribute__((address_space(3))) v4i*)l, 0, 0);
#else
  *(uint4*)l = *(const uint4*)g;
#endif
}
__device__ __forceinline__ void wait_fills() {
#if USE_ASYNC_LDS
  __builtin_amdgcn_s_wait_asynccnt(0);
#endif
}

// -------------------------------------------------------------------------
// One-time conversions: fp32 -> bf16 (activations), fp32 -> bf16 transposed
// (weights, stored as W^T[n][k] so GEMM B tiles are straight row copies).
// -------------------------------------------------------------------------
__global__ __launch_bounds__(256) void cvt_bf16(const float* __restrict__ in,
                                                unsigned short* __restrict__ out,
                                                int n8) {
  int i = blockIdx.x * 256 + threadIdx.x;
  if (i >= n8) return;
  const float4* p = (const float4*)in + (size_t)i * 2;
  float4 a = p[0], b = p[1];
  uint4 o;
  o.x = pack2(a.x, a.y); o.y = pack2(a.z, a.w);
  o.z = pack2(b.x, b.y); o.w = pack2(b.z, b.w);
  ((uint4*)out)[i] = o;
}

__global__ __launch_bounds__(128) void cvt_w_t(const float* __restrict__ W,
                                               unsigned short* __restrict__ Wt) {
  __shared__ unsigned short T[64][65];
  const int k0 = blockIdx.y << 6, n0 = blockIdx.x << 6;
  #pragma unroll
  for (int t = 0; t < 32; t++) {
    int i = threadIdx.x + (t << 7);
    int r = i >> 6, c = i & 63;                 // r = k-local, c = n-local
    T[r][c] = f2bf(W[(size_t)(k0 + r) * DIM + n0 + c]);
  }
  __syncthreads();
  #pragma unroll
  for (int t = 0; t < 32; t++) {
    int i = threadIdx.x + (t << 7);
    int r = i >> 6, c = i & 63;                 // r = n-local, c = k-local
    Wt[(size_t)(n0 + r) * DIM + k0 + c] = T[c][r];
  }
}

// -------------------------------------------------------------------------
// GEMM: out[M,3072] = A[M,3072] @ W[3072,3072] + bias, all-bf16 operands.
//   A:  bf16 row-major.  Wt: bf16 W^T, row-major in k.
//   mode 0: fp32 row-major into outF
//   mode 1: bf16 into outVt[h][d][S_TOT] (transposed V), s = s_off + row
// 128x64 block tile, 4 waves; wave computes 32x64 (8 WMMA / k-step).
// Double-buffered LDS, async global->LDS fills.
// -------------------------------------------------------------------------
__device__ __forceinline__ void fill_tiles(
    const unsigned short* __restrict__ A, const unsigned short* __restrict__ Wt,
    unsigned short (*As)[KS], unsigned short (*Bs)[KS],
    int tid, int m0, int n0, int kb)
{
  #pragma unroll
  for (int t = 0; t < 4; t++) {                 // A: 128x32 = 512 x 16B chunks
    int i = tid + (t << 7);
    int r = i >> 2, c = (i & 3) << 3;
    cp16(A + (size_t)(m0 + r) * DIM + kb + c, &As[r][c]);
  }
  #pragma unroll
  for (int t = 0; t < 2; t++) {                 // B: 64x32 = 256 x 16B chunks
    int i = tid + (t << 7);
    int r = i >> 2, c = (i & 3) << 3;
    cp16(Wt + (size_t)(n0 + r) * DIM + kb + c, &Bs[r][c]);
  }
}

__global__ __launch_bounds__(128) void gemm_bf16wmma(
    const unsigned short* __restrict__ A,
    const unsigned short* __restrict__ Wt,
    const float* __restrict__ bias,
    float* __restrict__ outF, unsigned short* __restrict__ outVt,
    int mode, int s_off)
{
  __shared__ __align__(16) unsigned short As[2][MT][KS];   // 16 KB
  __shared__ __align__(16) unsigned short Bs[2][NT][KS];   //  8 KB

  const int tid  = threadIdx.x;
  const int wv   = tid >> 5;
  const int lane = tid & 31;
  const int nl   = lane & 15;
  const int hh   = lane >> 4;
  const int m0   = blockIdx.y * MT;
  const int n0   = blockIdx.x * NT;

  f32x8 acc[2][4] = {};

  fill_tiles(A, Wt, As[0], Bs[0], tid, m0, n0, 0);

  for (int kb = 0; kb < DIM; kb += KS) {
    const int p = (kb >> 5) & 1;
    wait_fills();
    __syncthreads();
    if (kb + KS < DIM)
      fill_tiles(A, Wt, As[p ^ 1], Bs[p ^ 1], tid, m0, n0, kb + KS);

    // A fragments (16x32): lane half selects K [0,16)/[16,32)
    Frag16 a0, a1;
    #pragma unroll
    for (int i = 0; i < 8; i++) {
      int kk = (hh << 4) + (i << 1);
      a0.ui[i] = *(const unsigned int*)&As[p][(wv << 5) + nl][kk];
      a1.ui[i] = *(const unsigned int*)&As[p][(wv << 5) + 16 + nl][kk];
    }
    #pragma unroll
    for (int j = 0; j < 4; j++) {
      // B fragment (32x16): lane half selects K [0,16)/[16,32), 2 K per VGPR
      Frag16 b;
      #pragma unroll
      for (int i = 0; i < 8; i++)
        b.ui[i] = *(const unsigned int*)&Bs[p][(j << 4) + nl][(hh << 4) + (i << 1)];
      acc[0][j] = __builtin_amdgcn_wmma_f32_16x16x32_bf16(
          false, a0.v, false, b.v, (short)0, acc[0][j], false, false);
      acc[1][j] = __builtin_amdgcn_wmma_f32_16x16x32_bf16(
          false, a1.v, false, b.v, (short)0, acc[1][j], false, false);
    }
  }

  // C/D layout: VGPR r holds M = r + 8*hh, lane%16 = N
  #pragma unroll
  for (int mi = 0; mi < 2; mi++) {
    const int rowbase = m0 + (wv << 5) + (mi << 4) + (hh << 3);
    #pragma unroll
    for (int j = 0; j < 4; j++) {
      int ng = n0 + (j << 4) + nl;
      float bvl = bias[ng];
      #pragma unroll
      for (int r = 0; r < 8; r++) {
        float val = acc[mi][j][r] + bvl;
        int row = rowbase + r;
        if (mode == 0) {
          outF[(size_t)row * DIM + ng] = val;
        } else {
          int hd = ng >> 7, dd = ng & 127;
          outVt[((size_t)hd * HD + dd) * S_TOT + s_off + row] = f2bf(val);
        }
      }
    }
  }
}

// -------------------------------------------------------------------------
// Per-(s,h) RMSNorm + RoPE, one wave per (s,h); writes bf16 [H][S_TOT][HD].
// -------------------------------------------------------------------------
__global__ __launch_bounds__(128) void norm_rope(
    const float* __restrict__ in,                        // [S_TOT][DIM]
    const float* __restrict__ g_txt, const float* __restrict__ g_img,
    const float* __restrict__ tc, const float* __restrict__ ts,
    const float* __restrict__ ic, const float* __restrict__ isn,
    unsigned short* __restrict__ out)                    // [H][S_TOT][HD]
{
  const int lane = threadIdx.x & 31;
  const int idx = (blockIdx.x << 2) + (threadIdx.x >> 5);
  const int s = idx / HEADS, h = idx % HEADS;
  const float* row = in + (size_t)s * DIM + h * HD;
  const int d = lane << 2;
  float x0 = row[d], x1 = row[d + 1], x2 = row[d + 2], x3 = row[d + 3];
  float ss = x0 * x0 + x1 * x1 + x2 * x2 + x3 * x3;
  #pragma unroll
  for (int off = 1; off < 32; off <<= 1) ss += __shfl_xor(ss, off, 32);
  float inv = rsqrtf(ss * (1.0f / 128.0f) + 1e-6f);
  const float* g  = (s < S_TXT) ? g_txt : g_img;
  const float* ct = (s < S_TXT) ? tc + (size_t)s * 64 : ic + (size_t)(s - S_TXT) * 64;
  const float* st = (s < S_TXT) ? ts + (size_t)s * 64 : isn + (size_t)(s - S_TXT) * 64;
  float a0 = x0 * inv * g[d],     a1 = x1 * inv * g[d + 1];
  float a2 = x2 * inv * g[d + 2], a3 = x3 * inv * g[d + 3];
  float c0 = ct[d >> 1], s0 = st[d >> 1];
  float c1 = ct[(d >> 1) + 1], s1 = st[(d >> 1) + 1];
  uint2 o;
  o.x = pack2(a0 * c0 - a1 * s0, a0 * s0 + a1 * c0);
  o.y = pack2(a2 * c1 - a3 * s1, a2 * s1 + a3 * c1);
  *(uint2*)(out + ((size_t)h * S_TOT + s) * HD + d) = o;
}

// -------------------------------------------------------------------------
// Flash attention: block = (head, 64 queries); 4 waves, 16 q-rows each.
// kv blocks of 32; QK^T and PV via bf16 WMMA; online softmax in fp32.
// -------------------------------------------------------------------------
__global__ __launch_bounds__(128) void flash_attn(
    const unsigned short* __restrict__ Q,   // [H][S_TOT][HD]
    const unsigned short* __restrict__ K,   // [H][S_TOT][HD]
    const unsigned short* __restrict__ Vt,  // [H][HD][S_TOT]
    unsigned short* __restrict__ joint)     // [S_TOT][DIM]
{
  __shared__ __align__(16) unsigned short P[4][16][32];  // per-wave P patch
  const int h  = blockIdx.y;
  const int q0 = blockIdx.x << 6;
  const int wv   = threadIdx.x >> 5;
  const int lane = threadIdx.x & 31;
  const int nl   = lane & 15;
  const int hh   = lane >> 4;
  const float scale = 0.08838834764831845f;              // 1/sqrt(128)

  const unsigned short* Qh = Q  + (size_t)h * S_TOT * HD;
  const unsigned short* Kh = K  + (size_t)h * S_TOT * HD;
  const unsigned short* Vh = Vt + (size_t)h * HD * S_TOT;

  Frag16 qf[4];
  {
    const unsigned short* qr = Qh + (size_t)(q0 + (wv << 4) + nl) * HD;
    #pragma unroll
    for (int c = 0; c < 4; c++)
      #pragma unroll
      for (int i = 0; i < 8; i++) {
        int kk = (c << 5) + (hh << 4) + (i << 1);
        qf[c].ui[i] = *(const unsigned int*)(qr + kk);
      }
  }

  f32x8 o[8] = {};
  float m_run[8], l_run[8];
  #pragma unroll
  for (int r = 0; r < 8; r++) { m_run[r] = -3.0e38f; l_run[r] = 0.0f; }

  for (int k0 = 0; k0 < S_TOT; k0 += 32) {
    f32x8 sc[2] = {};
    #pragma unroll
    for (int j = 0; j < 2; j++) {
      const unsigned short* kr = Kh + (size_t)(k0 + (j << 4) + nl) * HD;
      #pragma unroll
      for (int c = 0; c < 4; c++) {
        Frag16 kfr;
        #pragma unroll
        for (int i = 0; i < 8; i++)
          kfr.ui[i] = *(const unsigned int*)(kr + (c << 5) + (hh << 4) + (i << 1));
        sc[j] = __builtin_amdgcn_wmma_f32_16x16x32_bf16(
            false, qf[c].v, false, kfr.v, (short)0, sc[j], false, false);
      }
    }
    #pragma unroll
    for (int r = 0; r < 8; r++) {
      float s0 = sc[0][r] * scale;
      float s1 = sc[1][r] * scale;
      float mx = fmaxf(s0, s1);
      #pragma unroll
      for (int off = 1; off < 16; off <<= 1) mx = fmaxf(mx, __shfl_xor(mx, off, 32));
      float mn = fmaxf(m_run[r], mx);
      float alpha = __expf(m_run[r] - mn);
      m_run[r] = mn;
      float p0 = __expf(s0 - mn);
      float p1 = __expf(s1 - mn);
      float rs = p0 + p1;
      #pragma unroll
      for (int off = 1; off < 16; off <<= 1) rs += __shfl_xor(rs, off, 32);
      l_run[r] = l_run[r] * alpha + rs;
      #pragma unroll
      for (int n = 0; n < 8; n++) o[n][r] *= alpha;
      int mrow = (hh << 3) + r;
      P[wv][mrow][nl]      = f2bf(p0);
      P[wv][mrow][16 + nl] = f2bf(p1);
    }
    Frag16 pf;
    #pragma unroll
    for (int i = 0; i < 8; i++) {
      int kk = (hh << 4) + (i << 1);
      pf.ui[i] = *(const unsigned int*)&P[wv][nl][kk];
    }
    #pragma unroll
    for (int n = 0; n < 8; n++) {
      Frag16 vf;
      const unsigned short* vr = Vh + (size_t)((n << 4) + nl) * S_TOT + k0;
      #pragma unroll
      for (int i = 0; i < 8; i++)
        vf.ui[i] = *(const unsigned int*)(vr + (hh << 4) + (i << 1));
      o[n] = __builtin_amdgcn_wmma_f32_16x16x32_bf16(
          false, pf.v, false, vf.v, (short)0, o[n], false, false);
    }
  }

  #pragma unroll
  for (int r = 0; r < 8; r++) {
    float invl = 1.0f / l_run[r];
    int row = q0 + (wv << 4) + (hh << 3) + r;
    unsigned short* jr = joint + (size_t)row * DIM + h * HD;
    #pragma unroll
    for (int n = 0; n < 8; n++)
      jr[(n << 4) + nl] = f2bf(o[n][r] * invl);
  }
}

// -------------------------------------------------------------------------
extern "C" void kernel_launch(void* const* d_in, const int* in_sizes, int n_in,
                              void* d_out, int out_size, void* d_ws, size_t ws_size,
                              hipStream_t stream) {
  (void)in_sizes; (void)n_in; (void)out_size; (void)ws_size;
  const float* image   = (const float*)d_in[0];
  const float* text    = (const float*)d_in[1];
  const float* img_cos = (const float*)d_in[2];
  const float* img_sin = (const float*)d_in[3];
  const float* txt_cos = (const float*)d_in[4];
  const float* txt_sin = (const float*)d_in[5];
  const float* bq  = (const float*)d_in[7];
  const float* bk  = (const float*)d_in[9];
  const float* bv  = (const float*)d_in[11];
  const float* baq = (const float*)d_in[13];
  const float* bak = (const float*)d_in[15];
  const float* bav = (const float*)d_in[17];
  const float* gq  = (const float*)d_in[18]; const float* gk  = (const float*)d_in[19];
  const float* gaq = (const float*)d_in[20]; const float* gak = (const float*)d_in[21];
  const float* bo  = (const float*)d_in[23];
  const float* bao = (const float*)d_in[25];
  const float* Wsrc[8] = {
      (const float*)d_in[6],  (const float*)d_in[8],  (const float*)d_in[10],
      (const float*)d_in[12], (const float*)d_in[14], (const float*)d_in[16],
      (const float*)d_in[22], (const float*)d_in[24]};  // Wq Wk Wv Waq Wak Wav Wo Wao

  // workspace carve-up (~293 MB)
  const size_t WSZ = (size_t)DIM * DIM;
  unsigned short* wbuf = (unsigned short*)d_ws;
  unsigned short* Wt[8];
  for (int i = 0; i < 8; i++) Wt[i] = wbuf + (size_t)i * WSZ;   // bf16 W^T
  unsigned short* imgb = wbuf + 8 * WSZ;                        // bf16 image
  unsigned short* txtb = imgb + (size_t)S_IMG * DIM;            // bf16 text
  float* qf = (float*)(txtb + (size_t)S_TXT * DIM);             // fp32 pre-norm q
  float* kf = qf + (size_t)S_TOT * DIM;                         // fp32 pre-norm k
  unsigned short* Qb  = (unsigned short*)(kf + (size_t)S_TOT * DIM);
  unsigned short* Kb  = Qb + (size_t)HEADS * S_TOT * HD;
  unsigned short* Vtb = Kb + (size_t)HEADS * S_TOT * HD;        // [H][D][S]
  unsigned short* joint = Vtb + (size_t)HEADS * S_TOT * HD;     // bf16 [S][DIM]

  dim3 blk(128);

  // one-time conversions
  for (int i = 0; i < 8; i++)
    cvt_w_t<<<dim3(DIM / 64, DIM / 64), blk, 0, stream>>>(Wsrc[i], Wt[i]);
  cvt_bf16<<<dim3(S_IMG * DIM / 8 / 256), 256, 0, stream>>>(image, imgb, S_IMG * DIM / 8);
  cvt_bf16<<<dim3(S_TXT * DIM / 8 / 256), 256, 0, stream>>>(text, txtb, S_TXT * DIM / 8);

  // QKV projections (text rows [0,512), image rows [512,2560))
  dim3 gi(DIM / NT, S_IMG / MT), gt(DIM / NT, S_TXT / MT);
  gemm_bf16wmma<<<gi, blk, 0, stream>>>(imgb, Wt[0], bq,
                                        qf + (size_t)S_TXT * DIM, nullptr, 0, 0);
  gemm_bf16wmma<<<gi, blk, 0, stream>>>(imgb, Wt[1], bk,
                                        kf + (size_t)S_TXT * DIM, nullptr, 0, 0);
  gemm_bf16wmma<<<gi, blk, 0, stream>>>(imgb, Wt[2], bv, nullptr, Vtb, 1, S_TXT);
  gemm_bf16wmma<<<gt, blk, 0, stream>>>(txtb, Wt[3], baq, qf, nullptr, 0, 0);
  gemm_bf16wmma<<<gt, blk, 0, stream>>>(txtb, Wt[4], bak, kf, nullptr, 0, 0);
  gemm_bf16wmma<<<gt, blk, 0, stream>>>(txtb, Wt[5], bav, nullptr, Vtb, 1, 0);

  dim3 gn(S_TOT * HEADS / 4);
  norm_rope<<<gn, blk, 0, stream>>>(qf, gaq, gq, txt_cos, txt_sin,
                                    img_cos, img_sin, Qb);
  norm_rope<<<gn, blk, 0, stream>>>(kf, gak, gk, txt_cos, txt_sin,
                                    img_cos, img_sin, Kb);

  flash_attn<<<dim3(S_TOT / 64, HEADS), blk, 0, stream>>>(Qb, Kb, Vtb, joint);

  // output projections: d_out = [img_out (2048x3072) | txt_out (512x3072)] fp32
  float* img_out = (float*)d_out;
  float* txt_out = img_out + (size_t)S_IMG * DIM;
  gemm_bf16wmma<<<gi, blk, 0, stream>>>(joint + (size_t)S_TXT * DIM, Wt[6], bo,
                                        img_out, nullptr, 0, 0);
  gemm_bf16wmma<<<gt, blk, 0, stream>>>(joint, Wt[7], bao, txt_out, nullptr, 0, 0);
}
